// PixelSequenceClassifier_39565238731519
// MI455X (gfx1250) — matrix operand order repaired and checked by reading, not compiled, not run on hardware
//
#include <hip/hip_runtime.h>
#include <hip/hip_bf16.h>

// ---------------------------------------------------------------------------
// Mamba pixel-sequence classifier for MI455X (gfx1250, wave32, WMMA).
// B=256 L=784 DM=64 DI=128 DS=16 DR=4 DC=4 NL=3 NC=10
//
// GEMMs use V_WMMA_F32_16X16X32_F16 with:
//   - weights pre-transposed to N x K so B fragments are contiguous v16h loads
//   - one wave per M-tile, A fragments held in registers across all N-tiles
// ---------------------------------------------------------------------------

typedef __attribute__((ext_vector_type(16))) _Float16 v16h;
typedef __attribute__((ext_vector_type(8)))  _Float16 v8h;
typedef __attribute__((ext_vector_type(8)))  float    v8f;

#define Bsz 256
#define Lsz 784
#define DM  64
#define DI  128
#define DS  16
#define DR  4
#define DCV 4
#define NL  3
#define NC  10

// ---------------- WMMA fragment helpers (C/D per ISA 7.12.2; A/B in contiguous k = 16h + i order) ----------------

// A: 16x32 f16 tile from row-major A (ld = ldA). lane%16 = row, lane/16 picks
// K chunks 0-15 vs 16-31 (element i <-> K=kb+i). Two contiguous 16B loads per lane.
__device__ __forceinline__ v16h load_frag_a(const _Float16* __restrict__ A, int ldA,
                                            int m0, int k0, int lane) {
    int row = m0 + (lane & 15);
    int kb  = k0 + ((lane >> 4) << 4);
    const _Float16* p = A + (size_t)row * ldA + kb;
    v8h lo = *(const v8h*)(p);
    v8h hi = *(const v8h*)(p + 8);
    v16h a;
#pragma unroll
    for (int i = 0; i < 8; ++i) { a[i] = lo[i]; a[i + 8] = hi[i]; }
    return a;
}

// B: 32x16 f16 tile from TRANSPOSED weights Wt (N x K row-major, ld = ldK).
// lane%16 = col n, lane/16 picks K rows 0-15 vs 16-31: element i <-> K=kb+i,
// which is exactly Wt[n][kb..kb+15] -> one contiguous 32B v16h load.
__device__ __forceinline__ v16h load_frag_bT(const _Float16* __restrict__ Wt, int ldK,
                                             int k0, int n0, int lane) {
    int col = n0 + (lane & 15);
    int kb  = k0 + ((lane >> 4) << 4);
    return *(const v16h*)(Wt + (size_t)col * ldK + kb);
}

// C/D: 16x16 f32 tile. VGPR r -> row r + 8*(lane/16), col = lane%16.
__device__ __forceinline__ v8f load_frag_c(const float* __restrict__ C, int ldC,
                                           int m0, int n0, int lane) {
    int col = n0 + (lane & 15);
    int rb  = m0 + ((lane >> 4) << 3);
    v8f c;
#pragma unroll
    for (int r = 0; r < 8; ++r) c[r] = C[(size_t)(rb + r) * ldC + col];
    return c;
}

__device__ __forceinline__ void store_frag_c(float* __restrict__ C, int ldC,
                                             int m0, int n0, int lane, v8f c) {
    int col = n0 + (lane & 15);
    int rb  = m0 + ((lane >> 4) << 3);
#pragma unroll
    for (int r = 0; r < 8; ++r) C[(size_t)(rb + r) * ldC + col] = c[r];
}

// ---------------- weight conversion (f32 -> f16, transposed to N x K) -------

// src (NL, K, N) f32 -> dst (NL, N, K) f16
__global__ void k_convert_T(const float* __restrict__ src, _Float16* __restrict__ dst,
                            int K, int N, int total) {
    int i = blockIdx.x * blockDim.x + threadIdx.x;
    if (i >= total) return;
    int kn = K * N;
    int l = i / kn, r = i % kn;
    int k = r / N,  n = r % N;
    dst[l * kn + n * K + k] = (_Float16)src[i];
}

// W_xproj (NL,128,36) f32 -> dst (NL,48,128) f16, pad rows (n>=36) zeroed.
__global__ void k_convert_xprojT(const float* __restrict__ src, _Float16* __restrict__ dst,
                                 int total) {
    int i = blockIdx.x * blockDim.x + threadIdx.x;
    if (i >= total) return;
    int l = i / (48 * DI);
    int r = i % (48 * DI);
    int n = r / DI;
    int k = r % DI;
    dst[i] = (n < 36) ? (_Float16)src[((size_t)l * DI + k) * 36 + n] : (_Float16)0.0f;
}

// ---------------- embedding ----------------

__global__ void k_embed(const float* __restrict__ x, const float* __restrict__ we,
                        const float* __restrict__ be, const float* __restrict__ pe,
                        float* __restrict__ H, long n) {
    long i = (long)blockIdx.x * blockDim.x + threadIdx.x;
    if (i >= n) return;
    int  d  = (int)(i & (DM - 1));
    long bl = i >> 6;
    int  l  = (int)(bl % Lsz);
    H[i] = x[bl] * we[d] + be[d] + pe[(long)l * DM + d];
}

// ---------------- layernorm (wave per row, wave32 shuffle reduce) ----------------

__global__ __launch_bounds__(256) void k_layernorm(const float* __restrict__ X,
                                                   const float* __restrict__ w,
                                                   const float* __restrict__ bvec,
                                                   _Float16* __restrict__ Y, long M) {
    int  lane = threadIdx.x & 31;
    int  wave = threadIdx.x >> 5;
    long row  = (long)blockIdx.x * 8 + wave;
    if (row >= M) return;
    const float* x = X + row * DM;
    float x0 = x[2 * lane], x1 = x[2 * lane + 1];
    float s = x0 + x1;
#pragma unroll
    for (int off = 16; off > 0; off >>= 1) s += __shfl_xor(s, off, 32);
    float mean = s * (1.0f / DM);
    float d0 = x0 - mean, d1 = x1 - mean;
    float v = d0 * d0 + d1 * d1;
#pragma unroll
    for (int off = 16; off > 0; off >>= 1) v += __shfl_xor(v, off, 32);
    float inv = rsqrtf(v * (1.0f / DM) + 1e-5f);
    _Float16* y = Y + row * DM;
    y[2 * lane]     = (_Float16)(d0 * inv * w[2 * lane]     + bvec[2 * lane]);
    y[2 * lane + 1] = (_Float16)(d1 * inv * w[2 * lane + 1] + bvec[2 * lane + 1]);
}

// ---------------- GEMM 1: xz = ln(u) @ W_in  (M x 64) @ (64 x 256) ----------------
// One wave per M-tile; A fragments reused across all 16 N-tiles.
// Columns 0..127 -> XP, 128..255 -> Z.

__global__ __launch_bounds__(256) void k_gemm_in(const _Float16* __restrict__ A,
                                                 const _Float16* __restrict__ Wt, // 256 x 64
                                                 float* __restrict__ XP,
                                                 float* __restrict__ Z, long Mtiles) {
    int  lane = threadIdx.x & 31;
    int  wave = threadIdx.x >> 5;
    long tM   = (long)blockIdx.x * 8 + wave;
    if (tM >= Mtiles) return;                 // wave-uniform: EXEC stays all-1s
    int m0 = (int)tM * 16;
    v16h a0 = load_frag_a(A, DM, m0, 0,  lane);
    v16h a1 = load_frag_a(A, DM, m0, 32, lane);
    for (int tN = 0; tN < 16; ++tN) {
        v16h b0 = load_frag_bT(Wt, DM, 0,  tN * 16, lane);
        v16h b1 = load_frag_bT(Wt, DM, 32, tN * 16, lane);
        v8f c = {};
        c = __builtin_amdgcn_wmma_f32_16x16x32_f16(false, a0, false, b0, (short)0, c, false, false);
        c = __builtin_amdgcn_wmma_f32_16x16x32_f16(false, a1, false, b1, (short)0, c, false, false);
        if (tN < 8) store_frag_c(XP, DI, m0, tN * 16, lane, c);
        else        store_frag_c(Z,  DI, m0, (tN - 8) * 16, lane, c);
    }
}

// ---------------- depthwise causal conv (DC=4) + bias + SiLU ----------------

__global__ void k_conv_silu(const float* __restrict__ XP, const float* __restrict__ cw,
                            const float* __restrict__ cb, float* __restrict__ XA,
                            _Float16* __restrict__ XAH, long n) {
    long i = (long)blockIdx.x * blockDim.x + threadIdx.x;
    if (i >= n) return;
    int  d  = (int)(i & (DI - 1));
    long bl = i >> 7;
    int  l  = (int)(bl % Lsz);
    float acc = cb[d];
#pragma unroll
    for (int j = 0; j < DCV; ++j) {
        int lj = l - (DCV - 1) + j;
        if (lj >= 0) acc += XP[(bl - (DCV - 1) + j) * DI + d] * cw[d * DCV + j];
    }
    float s = acc * (1.0f / (1.0f + __expf(-acc)));
    XA[i]  = s;
    XAH[i] = (_Float16)s;
}

// ---------------- GEMM 2: prm = xp @ W_xproj  (M x 128) @ (128 x 48pad) ----------------

__global__ __launch_bounds__(256) void k_gemm_xproj(const _Float16* __restrict__ A,
                                                    const _Float16* __restrict__ Wt, // 48 x 128
                                                    float* __restrict__ PRM, long Mtiles) {
    int  lane = threadIdx.x & 31;
    int  wave = threadIdx.x >> 5;
    long tM   = (long)blockIdx.x * 8 + wave;
    if (tM >= Mtiles) return;
    int m0 = (int)tM * 16;
    v16h a[4];
#pragma unroll
    for (int ks = 0; ks < 4; ++ks) a[ks] = load_frag_a(A, DI, m0, ks * 32, lane);
#pragma unroll
    for (int tN = 0; tN < 3; ++tN) {
        v8f c = {};
#pragma unroll
        for (int ks = 0; ks < 4; ++ks) {
            v16h b = load_frag_bT(Wt, DI, ks * 32, tN * 16, lane);
            c = __builtin_amdgcn_wmma_f32_16x16x32_f16(false, a[ks], false, b, (short)0, c, false, false);
        }
        store_frag_c(PRM, 48, m0, tN * 16, lane, c);
    }
}

// ---------------- dt = softplus(prm[:, :4] @ W_dt + b_dt) (K=4: VALU) ----------------

__global__ void k_dt(const float* __restrict__ PRM, const float* __restrict__ Wdt,
                     const float* __restrict__ bdt, float* __restrict__ DT, long n) {
    long i = (long)blockIdx.x * blockDim.x + threadIdx.x;
    if (i >= n) return;
    int  d   = (int)(i & (DI - 1));
    long row = i >> 7;
    const float* p = PRM + row * 48;
    float acc = bdt[d];
#pragma unroll
    for (int r = 0; r < DR; ++r) acc += p[r] * Wdt[r * DI + d];
    DT[i] = (acc > 20.0f) ? acc : log1pf(__expf(acc));
}

// ---------------- selective scan: thread = (b,d), block = one batch row ----------------
// h_s <- exp(dt*A_s)*h_s + dt*x*B_s ; y = sum h_s*C_s ; fused +x*D and *silu(z).

__global__ __launch_bounds__(128) void k_scan(const float* __restrict__ DT,
                                              const float* __restrict__ XA,
                                              const float* __restrict__ PRM,
                                              const float* __restrict__ Z,
                                              const float* __restrict__ Alog,
                                              const float* __restrict__ Dp,
                                              _Float16* __restrict__ Y) {
    int b = blockIdx.x;
    int d = threadIdx.x;
    __shared__ float Bs[DS];
    __shared__ float Cs[DS];
    float A[DS];
#pragma unroll
    for (int s = 0; s < DS; ++s) A[s] = -__expf(Alog[d * DS + s]);
    float Dv = Dp[d];
    float h[DS];
#pragma unroll
    for (int s = 0; s < DS; ++s) h[s] = 0.0f;
    long base = (long)b * Lsz;
    for (int l = 0; l < Lsz; ++l) {
        long bl = base + l;
        float dt = DT[bl * DI + d];
        float xt = XA[bl * DI + d];
        float zt = Z[bl * DI + d];
        if (l + 1 < Lsz) __builtin_prefetch(&DT[(bl + 1) * DI + d], 0, 1);
        if (d < DS)            Bs[d]      = PRM[bl * 48 + DR + d];
        else if (d < 2 * DS)   Cs[d - DS] = PRM[bl * 48 + DR + DS + (d - DS)];
        __syncthreads();
        float dtx = dt * xt;
        float y = 0.0f;
#pragma unroll
        for (int s = 0; s < DS; ++s) {
            float dA = __expf(dt * A[s]);
            h[s] = dA * h[s] + dtx * Bs[s];
            y += h[s] * Cs[s];
        }
        y = (y + xt * Dv) * (zt / (1.0f + __expf(-zt)));
        Y[bl * DI + d] = (_Float16)y;
        __syncthreads();
    }
}

// ---------------- GEMM 3: h += y @ W_out  (M x 128) @ (128 x 64), C preloaded ----------------

__global__ __launch_bounds__(256) void k_gemm_out(const _Float16* __restrict__ A,
                                                  const _Float16* __restrict__ Wt, // 64 x 128
                                                  float* __restrict__ H, long Mtiles) {
    int  lane = threadIdx.x & 31;
    int  wave = threadIdx.x >> 5;
    long tM   = (long)blockIdx.x * 8 + wave;
    if (tM >= Mtiles) return;
    int m0 = (int)tM * 16;
    v16h a[4];
#pragma unroll
    for (int ks = 0; ks < 4; ++ks) a[ks] = load_frag_a(A, DI, m0, ks * 32, lane);
#pragma unroll
    for (int tN = 0; tN < 4; ++tN) {
        v8f c = load_frag_c(H, DM, m0, tN * 16, lane);   // residual
#pragma unroll
        for (int ks = 0; ks < 4; ++ks) {
            v16h b = load_frag_bT(Wt, DI, ks * 32, tN * 16, lane);
            c = __builtin_amdgcn_wmma_f32_16x16x32_f16(false, a[ks], false, b, (short)0, c, false, false);
        }
        store_frag_c(H, DM, m0, tN * 16, lane, c);
    }
}

// ---------------- head: mean-pool over L, layernorm, classify ----------------

__global__ __launch_bounds__(64) void k_head(const float* __restrict__ H,
                                             const float* __restrict__ nw,
                                             const float* __restrict__ nb,
                                             const float* __restrict__ Wc,
                                             const float* __restrict__ bc,
                                             float* __restrict__ OUT) {
    int b = blockIdx.x;
    int d = threadIdx.x;
    __shared__ float red[DM];
    __shared__ float sp[DM];
    float s = 0.0f;
    const float* hp = H + (long)b * Lsz * DM + d;
    for (int l = 0; l < Lsz; ++l) s += hp[(long)l * DM];
    float pooled = s * (1.0f / Lsz);
    red[d] = pooled;
    __syncthreads();
#pragma unroll
    for (int off = 32; off > 0; off >>= 1) { if (d < off) red[d] += red[d + off]; __syncthreads(); }
    float mean = red[0] * (1.0f / DM);
    __syncthreads();
    float diff = pooled - mean;
    red[d] = diff * diff;
    __syncthreads();
#pragma unroll
    for (int off = 32; off > 0; off >>= 1) { if (d < off) red[d] += red[d + off]; __syncthreads(); }
    float var = red[0] * (1.0f / DM);
    sp[d] = diff * rsqrtf(var + 1e-5f) * nw[d] + nb[d];
    __syncthreads();
    if (d < NC) {
        float acc = bc[d];
#pragma unroll
        for (int k = 0; k < DM; ++k) acc += sp[k] * Wc[k * NC + d];
        OUT[b * NC + d] = acc;
    }
}

// ---------------------------------------------------------------------------

extern "C" void kernel_launch(void* const* d_in, const int* in_sizes, int n_in,
                              void* d_out, int out_size, void* d_ws, size_t ws_size,
                              hipStream_t stream) {
    (void)in_sizes; (void)n_in; (void)out_size; (void)ws_size;

    const float* x         = (const float*)d_in[0];
    const float* W_embed   = (const float*)d_in[1];
    const float* b_embed   = (const float*)d_in[2];
    const float* pos_embed = (const float*)d_in[3];
    const float* ln_w      = (const float*)d_in[4];
    const float* ln_b      = (const float*)d_in[5];
    const float* W_in      = (const float*)d_in[6];
    const float* conv_w    = (const float*)d_in[7];
    const float* conv_b    = (const float*)d_in[8];
    const float* W_xproj   = (const float*)d_in[9];
    const float* W_dt      = (const float*)d_in[10];
    const float* b_dt      = (const float*)d_in[11];
    const float* A_log     = (const float*)d_in[12];
    const float* Dp        = (const float*)d_in[13];
    const float* W_out     = (const float*)d_in[14];
    const float* norm_w    = (const float*)d_in[15];
    const float* norm_b    = (const float*)d_in[16];
    const float* W_cls     = (const float*)d_in[17];
    const float* b_cls     = (const float*)d_in[18];
    float* out = (float*)d_out;

    const long M      = (long)Bsz * Lsz;        // 200704 token rows
    const long Mtiles = M / 16;                 // 12544 (divisible by 8)

    char*  ws  = (char*)d_ws;
    size_t off = 0;
    auto alloc = [&](size_t bytes) -> void* {
        void* p = ws + off;
        off += (bytes + 255) & ~(size_t)255;
        return p;
    };

    float*    H       = (float*)   alloc((size_t)M * DM * sizeof(float));
    _Float16* ULN     = (_Float16*)alloc((size_t)M * DM * sizeof(_Float16));
    float*    XP      = (float*)   alloc((size_t)M * DI * sizeof(float));
    float*    Zb      = (float*)   alloc((size_t)M * DI * sizeof(float));
    float*    XA      = (float*)   alloc((size_t)M * DI * sizeof(float));
    _Float16* XAH     = (_Float16*)alloc((size_t)M * DI * sizeof(_Float16));
    float*    PRM     = (float*)   alloc((size_t)M * 48 * sizeof(float));
    float*    DTb     = (float*)   alloc((size_t)M * DI * sizeof(float));
    _Float16* Yh      = (_Float16*)alloc((size_t)M * DI * sizeof(_Float16));
    _Float16* WIN16T  = (_Float16*)alloc((size_t)NL * 2 * DI * DM * sizeof(_Float16)); // (l, 256, 64)
    _Float16* WXP16T  = (_Float16*)alloc((size_t)NL * 48 * DI * sizeof(_Float16));     // (l, 48, 128)
    _Float16* WOUT16T = (_Float16*)alloc((size_t)NL * DM * DI * sizeof(_Float16));     // (l, 64, 128)

    // weight conversions + transposes (deterministic, redone each call)
    {
        int n1 = NL * DM * 2 * DI;
        k_convert_T<<<(n1 + 255) / 256, 256, 0, stream>>>(W_in, WIN16T, DM, 2 * DI, n1);
        int n2 = NL * DI * DM;
        k_convert_T<<<(n2 + 255) / 256, 256, 0, stream>>>(W_out, WOUT16T, DI, DM, n2);
        int n3 = NL * 48 * DI;
        k_convert_xprojT<<<(n3 + 255) / 256, 256, 0, stream>>>(W_xproj, WXP16T, n3);
    }

    // embedding
    {
        long n = M * DM;
        k_embed<<<(int)((n + 255) / 256), 256, 0, stream>>>(x, W_embed, b_embed, pos_embed, H, n);
    }

    for (int l = 0; l < NL; ++l) {
        k_layernorm<<<(int)(M / 8), 256, 0, stream>>>(H, ln_w + l * DM, ln_b + l * DM, ULN, M);

        k_gemm_in<<<(int)(Mtiles / 8), 256, 0, stream>>>(
            ULN, WIN16T + (size_t)l * 2 * DI * DM, XP, Zb, Mtiles);

        long nc = M * DI;
        k_conv_silu<<<(int)((nc + 255) / 256), 256, 0, stream>>>(
            XP, conv_w + (size_t)l * DI * DCV, conv_b + (size_t)l * DI, XA, XAH, nc);

        k_gemm_xproj<<<(int)(Mtiles / 8), 256, 0, stream>>>(
            XAH, WXP16T + (size_t)l * 48 * DI, PRM, Mtiles);

        k_dt<<<(int)((nc + 255) / 256), 256, 0, stream>>>(
            PRM, W_dt + (size_t)l * DR * DI, b_dt + (size_t)l * DI, DTb, nc);

        k_scan<<<Bsz, DI, 0, stream>>>(
            DTb, XA, PRM, Zb, A_log + (size_t)l * DI * DS, Dp + (size_t)l * DI, Yh);

        k_gemm_out<<<(int)(Mtiles / 8), 256, 0, stream>>>(
            Yh, WOUT16T + (size_t)l * DM * DI, H, Mtiles);
    }

    k_head<<<Bsz, DM, 0, stream>>>(H, norm_w, norm_b, W_cls, b_cls, out);
}
